// NeRFMoE_53403623358650
// MI455X (gfx1250) — matrix operand order repaired and checked by reading, not compiled, not run on hardware
//
#include <hip/hip_runtime.h>
#include <hip/hip_bf16.h>
#include <math.h>

// ---------------------------------------------------------------------------
// NeRF MoE forward on gfx1250 (CDNA5), wave32, WMMA f16->f32.
// ---------------------------------------------------------------------------

#define NTOK   65536
#define NPAD   (NTOK + 128)          // slab rows incl. per-expert 16-alignment pad
#define HDIM   256
#define EXP    8
#define LAYERS 4
#define HH     (HDIM * HDIM)
#define CXP    64                    // 63 posenc dims padded to 64
#define CINP   288                   // 256 + 27 padded to 288
#define HEADD  128

typedef _Float16 half_t;
typedef __attribute__((ext_vector_type(16))) _Float16 v16h;
typedef __attribute__((ext_vector_type(8)))  _Float16 v8h;
typedef __attribute__((ext_vector_type(8)))  float    v8f;

// ---------------------------------------------------------------------------
// Generic WMMA GEMM:  Y[M,Nc] = act( A[M,K] @ Bt[Nc,K]^T + bias + R )
//   A  : f16 row-major, lda = K
//   Bt : f16, transposed weights, row-major [Nc, K]
//   R  : optional f16 residual [M, Nc]
//   padOff (9 ints): expert-slab mode -- per 16-row M-tile pick expert e with
//   padOff[e] <= m0 < padOff[e+1]; weights at Bt + e*wstride, bias + e*bstride.
// Block: 128 threads = 4 waves; wave w -> rows [bx*64 + w*16, +16), cols
// [by*64, +64) as 4 accumulators; K-loop step 32 -> 4 v_wmma per iteration.
// ---------------------------------------------------------------------------
__global__ __launch_bounds__(128) void gemm_wmma_f16(
    const half_t* __restrict__ A,
    const half_t* __restrict__ Bt,
    const float*  __restrict__ bias,
    const half_t* __restrict__ R,
    half_t* __restrict__ Y,
    int K, int Ncols, int relu,
    const int* __restrict__ padOff, int wstride, int bstride)
{
  const int lane = threadIdx.x & 31;
  const int wave = threadIdx.x >> 5;
  const int m0   = blockIdx.x * 64 + wave * 16;
  const int n0   = blockIdx.y * 64;

  const half_t* Bw = Bt;
  const float*  bw = bias;
  if (padOff) {
    int e = 0;
#pragma unroll
    for (int i = 1; i < EXP; ++i)
      if (m0 >= padOff[i]) e = i;
    Bw += (size_t)e * (size_t)wstride;
    bw += (size_t)e * (size_t)bstride;
  }

  v8f acc[4] = {};

  // A-fragment addressing (contiguous K, matching B):
  //   lane<16 : row m0+lane,     K  = {k..k+15}
  //   lane>=16: row m0+lane-16,  K  = {k+16..k+31}
  const int hs   = lane >> 4;                 // half-select
  const int arow = m0 + (lane & 15);
  const half_t* Ar = A + (size_t)arow * K + hs * 16;

  // B-fragment: lane -> col n0+(lane&15); lane-half selects K 0..15 / 16..31.
  const int bcol = lane & 15;
  const int bk   = hs * 16;

  for (int k = 0; k < K; k += 32) {
    v8h alo = *(const v8h*)(Ar + k);
    v8h ahi = *(const v8h*)(Ar + k + 8);
    v16h a;
#pragma unroll
    for (int i = 0; i < 8; ++i) { a[i] = alo[i]; a[8 + i] = ahi[i]; }

#pragma unroll
    for (int j = 0; j < 4; ++j) {
      const half_t* Bp = Bw + (size_t)(n0 + j * 16 + bcol) * K + k + bk;
      v16h b = *(const v16h*)Bp;
      acc[j] = __builtin_amdgcn_wmma_f32_16x16x32_f16(
          false, a, false, b, (short)0, acc[j], false, false);
    }
  }

  // C/D layout: lane -> col (lane&15); VGPR v -> row m0 + v + (lane>>4)*8
  const int crow = m0 + (lane >> 4) * 8;
  const int ccol = lane & 15;
#pragma unroll
  for (int j = 0; j < 4; ++j) {
    const int col = n0 + j * 16 + ccol;
    const float bv = bw[col];
#pragma unroll
    for (int v = 0; v < 8; ++v) {
      const int row = crow + v;
      float x = acc[j][v] + bv;
      if (R) x += (float)R[(size_t)row * Ncols + col];
      if (relu) x = fmaxf(x, 0.0f);
      Y[(size_t)row * Ncols + col] = (half_t)x;
    }
  }
}

// ---------------------------------------------------------------------------
// Small helper kernels
// ---------------------------------------------------------------------------
__global__ void fill_i32_k(int* p, int v, int n) {
  int i = blockIdx.x * blockDim.x + threadIdx.x;
  if (i < n) p[i] = v;
}

// f32 [Kin, Nc] -> f16 transposed [Nc, Kpad] (K zero-padded)
__global__ void cvt_wT_k(const float* __restrict__ W, half_t* __restrict__ Wt,
                         int Kin, int Kpad, int Nc) {
  const int mat = blockIdx.z;
  W  += (size_t)mat * (size_t)Kin  * (size_t)Nc;
  Wt += (size_t)mat * (size_t)Kpad * (size_t)Nc;
  int idx = blockIdx.x * blockDim.x + threadIdx.x;
  if (idx >= Nc * Kpad) return;
  int n = idx / Kpad, k = idx % Kpad;
  float v = (k < Kin) ? W[(size_t)k * Nc + n] : 0.0f;
  Wt[(size_t)n * Kpad + k] = (half_t)v;
}

__global__ void posenc_xyz_k(const float* __restrict__ xyz,
                             half_t* __restrict__ ex) {
  int t = blockIdx.x * blockDim.x + threadIdx.x;
  if (t >= NTOK) return;
  float v0 = xyz[t * 3 + 0], v1 = xyz[t * 3 + 1], v2 = xyz[t * 3 + 2];
  half_t* row = ex + (size_t)t * CXP;
  row[0] = (half_t)v0; row[1] = (half_t)v1; row[2] = (half_t)v2;
  int o = 3;
  float f = 1.0f;
#pragma unroll
  for (int l = 0; l < 10; ++l) {
    row[o + 0] = (half_t)__sinf(v0 * f);
    row[o + 1] = (half_t)__sinf(v1 * f);
    row[o + 2] = (half_t)__sinf(v2 * f);
    row[o + 3] = (half_t)__cosf(v0 * f);
    row[o + 4] = (half_t)__cosf(v1 * f);
    row[o + 5] = (half_t)__cosf(v2 * f);
    o += 6; f *= 2.0f;
  }
  row[63] = (half_t)0.0f;   // pad 63 -> 64
}

// top-1 gate: logits = h @ Wg + bg, softmax score of argmax; count per expert
__global__ void gate_k(const half_t* __restrict__ h,
                       const float* __restrict__ Wg,
                       const float* __restrict__ bg,
                       int* __restrict__ top1, float* __restrict__ gatev,
                       int* __restrict__ cnt) {
  int t = blockIdx.x * blockDim.x + threadIdx.x;
  if (t >= NTOK) return;
  const half_t* hr = h + (size_t)t * HDIM;
  float lg[EXP];
#pragma unroll
  for (int e = 0; e < EXP; ++e) lg[e] = bg[e];
  for (int k = 0; k < HDIM; ++k) {
    float hv = (float)hr[k];
#pragma unroll
    for (int e = 0; e < EXP; ++e) lg[e] += hv * Wg[k * EXP + e];
  }
  float m = lg[0]; int am = 0;
#pragma unroll
  for (int e = 1; e < EXP; ++e)
    if (lg[e] > m) { m = lg[e]; am = e; }   // first-max wins, matches argmax
  float s = 0.0f;
#pragma unroll
  for (int e = 0; e < EXP; ++e) s += expf(lg[e] - m);
  top1[t]  = am;
  gatev[t] = expf(lg[am] - m) / s;
  atomicAdd(&cnt[am], 1);
}

// 16-aligned exclusive scan of per-expert counts; zero cursors
__global__ void scan_k(const int* __restrict__ cnt, int* __restrict__ padOff,
                       int* __restrict__ cursor) {
  if (threadIdx.x == 0) {
    int acc = 0;
#pragma unroll
    for (int e = 0; e < EXP; ++e) {
      padOff[e] = acc;
      acc += (cnt[e] + 15) & ~15;
      cursor[e] = 0;
    }
    padOff[EXP] = acc;
  }
}

__global__ void assign_k(const int* __restrict__ top1,
                         const int* __restrict__ padOff, int* __restrict__ cursor,
                         int* __restrict__ posA, int* __restrict__ perm) {
  int t = blockIdx.x * blockDim.x + threadIdx.x;
  if (t >= NTOK) return;
  int e   = top1[t];
  int idx = atomicAdd(&cursor[e], 1);
  int pos = padOff[e] + idx;
  posA[t]   = pos;
  perm[pos] = t;
}

// slab row <- token row of h (f16); padding rows (perm==-1) zeroed
__global__ void gather_k(const half_t* __restrict__ h,
                         const int* __restrict__ perm,
                         half_t* __restrict__ hg) {
  int row = blockIdx.x;
  int c   = threadIdx.x * 4;
  int t   = perm[row];
  half_t* dst = hg + (size_t)row * HDIM + c;
  if (t >= 0) {
    const half_t* src = h + (size_t)t * HDIM + c;
#pragma unroll
    for (int i = 0; i < 4; ++i) dst[i] = src[i];
  } else {
#pragma unroll
    for (int i = 0; i < 4; ++i) dst[i] = (half_t)0.0f;
  }
}

// moe_out[token] = gate[token] * tout[slab_pos(token)]
__global__ void scatter_k(const half_t* __restrict__ tout,
                          const int* __restrict__ posA,
                          const float* __restrict__ gatev,
                          half_t* __restrict__ moe) {
  int t = blockIdx.x;
  int c = threadIdx.x * 4;
  int pos = posA[t];
  float g = gatev[t];
  const half_t* src = tout + (size_t)pos * HDIM + c;
  half_t* dst = moe + (size_t)t * HDIM + c;
#pragma unroll
  for (int i = 0; i < 4; ++i) dst[i] = (half_t)(g * (float)src[i]);
}

__global__ void copy_moe_cin_k(const half_t* __restrict__ moe,
                               half_t* __restrict__ cin) {
  int t = blockIdx.x;
  int c = threadIdx.x * 4;
  const half_t* src = moe + (size_t)t * HDIM + c;
  half_t* dst = cin + (size_t)t * CINP + c;
#pragma unroll
  for (int i = 0; i < 4; ++i) dst[i] = src[i];
}

__global__ void posenc_dir_k(const float* __restrict__ dirs,
                             half_t* __restrict__ cin) {
  int t = blockIdx.x * blockDim.x + threadIdx.x;
  if (t >= NTOK) return;
  float v0 = dirs[t * 3 + 0], v1 = dirs[t * 3 + 1], v2 = dirs[t * 3 + 2];
  half_t* row = cin + (size_t)t * CINP + HDIM;   // cols 256..287
  row[0] = (half_t)v0; row[1] = (half_t)v1; row[2] = (half_t)v2;
  int o = 3;
  float f = 1.0f;
#pragma unroll
  for (int l = 0; l < 4; ++l) {
    row[o + 0] = (half_t)__sinf(v0 * f);
    row[o + 1] = (half_t)__sinf(v1 * f);
    row[o + 2] = (half_t)__sinf(v2 * f);
    row[o + 3] = (half_t)__cosf(v0 * f);
    row[o + 4] = (half_t)__cosf(v1 * f);
    row[o + 5] = (half_t)__cosf(v2 * f);
    o += 6; f *= 2.0f;
  }
#pragma unroll
  for (int i = 27; i < 32; ++i) row[i] = (half_t)0.0f;   // pad 283 -> 288
}

// sigma = softplus(s1 @ Ws2 + bs2); rgb = sigmoid(c1 @ Wc2 + bc2)
__global__ void heads_k(const half_t* __restrict__ s1,
                        const half_t* __restrict__ c1,
                        const float* __restrict__ Ws2, const float* __restrict__ bs2,
                        const float* __restrict__ Wc2, const float* __restrict__ bc2,
                        float* __restrict__ out) {
  int t = blockIdx.x * blockDim.x + threadIdx.x;
  if (t >= NTOK) return;
  const half_t* sr = s1 + (size_t)t * HEADD;
  const half_t* cr = c1 + (size_t)t * HEADD;
  float sig = bs2[0];
  float r0 = bc2[0], r1 = bc2[1], r2 = bc2[2];
  for (int k = 0; k < HEADD; ++k) {
    sig += (float)sr[k] * Ws2[k];
    float cv = (float)cr[k];
    r0 += cv * Wc2[k * 3 + 0];
    r1 += cv * Wc2[k * 3 + 1];
    r2 += cv * Wc2[k * 3 + 2];
  }
  float sp = (sig > 20.0f) ? sig : log1pf(expf(sig));
  out[t * 4 + 0] = 1.0f / (1.0f + expf(-r0));
  out[t * 4 + 1] = 1.0f / (1.0f + expf(-r1));
  out[t * 4 + 2] = 1.0f / (1.0f + expf(-r2));
  out[t * 4 + 3] = sp;
}

// ---------------------------------------------------------------------------
// Launcher
// ---------------------------------------------------------------------------
extern "C" void kernel_launch(void* const* d_in, const int* in_sizes, int n_in,
                              void* d_out, int out_size, void* d_ws, size_t ws_size,
                              hipStream_t stream) {
  const float* xyz    = (const float*)d_in[0];
  const float* dirs   = (const float*)d_in[1];
  const float* W_xyz1 = (const float*)d_in[2];
  const float* b_xyz1 = (const float*)d_in[3];
  const float* W_xyz2 = (const float*)d_in[4];
  const float* b_xyz2 = (const float*)d_in[5];
  const float* Wg     = (const float*)d_in[6];
  const float* bg     = (const float*)d_in[7];
  const float* We     = (const float*)d_in[8];
  const float* be     = (const float*)d_in[9];
  const float* Ws1    = (const float*)d_in[10];
  const float* bs1    = (const float*)d_in[11];
  const float* Ws2    = (const float*)d_in[12];
  const float* bs2    = (const float*)d_in[13];
  const float* Wc1    = (const float*)d_in[14];
  const float* bc1    = (const float*)d_in[15];
  const float* Wc2    = (const float*)d_in[16];
  const float* bc2    = (const float*)d_in[17];
  float* out = (float*)d_out;

  // ---- workspace carve (256B aligned) ----
  char*  ws  = (char*)d_ws;
  size_t cur = 0;
  auto alloc = [&](size_t bytes) -> void* {
    void* p = ws + cur;
    cur += (bytes + 255) & ~(size_t)255;
    return p;
  };
  half_t* ex   = (half_t*)alloc((size_t)NTOK * CXP * 2);
  half_t* h1   = (half_t*)alloc((size_t)NTOK * HDIM * 2);   // trunk mid; aliased by moe later
  half_t* hf   = (half_t*)alloc((size_t)NTOK * HDIM * 2);   // trunk out (f16)
  half_t* hg   = (half_t*)alloc((size_t)NPAD * HDIM * 2);   // expert slab input
  half_t* t1   = (half_t*)alloc((size_t)NPAD * HDIM * 2);
  half_t* t2   = (half_t*)alloc((size_t)NPAD * HDIM * 2);
  half_t* Wx1t = (half_t*)alloc((size_t)HDIM * CXP * 2);
  half_t* Wx2t = (half_t*)alloc((size_t)HDIM * HDIM * 2);
  half_t* Wet  = (half_t*)alloc((size_t)EXP * LAYERS * HH * 2);
  half_t* Ws1t = (half_t*)alloc((size_t)HEADD * HDIM * 2);
  half_t* Wc1t = (half_t*)alloc((size_t)HEADD * CINP * 2);
  int*   top1   = (int*)alloc((size_t)NTOK * 4);
  int*   posA   = (int*)alloc((size_t)NTOK * 4);
  int*   perm   = (int*)alloc((size_t)NPAD * 4);
  float* gatev  = (float*)alloc((size_t)NTOK * 4);
  int*   cnt    = (int*)alloc(64);
  int*   cursor = (int*)alloc(64);
  int*   padOff = (int*)alloc(64);

  // aliases over freed regions (safe by launch ordering on one stream):
  half_t* moe = h1;                              // used after trunk2 done
  half_t* cin = hg;                              // used after expert GEMMs done
  half_t* s1b = t2;                              // used after scatter done
  half_t* c1b = t2 + (size_t)NTOK * HEADD;       // fits within t2 region

  // ---- init ----
  fill_i32_k<<<1, 32, 0, stream>>>(cnt, 0, EXP);
  fill_i32_k<<<(NPAD + 255) / 256, 256, 0, stream>>>(perm, -1, NPAD);

  // ---- weight conversion (f32 -> f16 transposed, K-padded) ----
  auto cvt = [&](const float* W, half_t* Wt, int Kin, int Kpad, int Nc, int nmat) {
    int per = Nc * Kpad;
    dim3 g((per + 255) / 256, 1, nmat);
    cvt_wT_k<<<g, 256, 0, stream>>>(W, Wt, Kin, Kpad, Nc);
  };
  cvt(W_xyz1, Wx1t, 63, CXP, HDIM, 1);
  cvt(W_xyz2, Wx2t, HDIM, HDIM, HDIM, 1);
  cvt(We, Wet, HDIM, HDIM, HDIM, EXP * LAYERS);
  cvt(Ws1, Ws1t, HDIM, HDIM, HEADD, 1);
  cvt(Wc1, Wc1t, HDIM + 27, CINP, HEADD, 1);

  // ---- trunk ----
  posenc_xyz_k<<<NTOK / 256, 256, 0, stream>>>(xyz, ex);
  gemm_wmma_f16<<<dim3(NTOK / 64, HDIM / 64), 128, 0, stream>>>(
      ex, Wx1t, b_xyz1, nullptr, h1, CXP, HDIM, 1, nullptr, 0, 0);
  gemm_wmma_f16<<<dim3(NTOK / 64, HDIM / 64), 128, 0, stream>>>(
      h1, Wx2t, b_xyz2, nullptr, hf, HDIM, HDIM, 0, nullptr, 0, 0);

  // ---- routing ----
  gate_k<<<NTOK / 256, 256, 0, stream>>>(hf, Wg, bg, top1, gatev, cnt);
  scan_k<<<1, 1, 0, stream>>>(cnt, padOff, cursor);
  assign_k<<<NTOK / 256, 256, 0, stream>>>(top1, padOff, cursor, posA, perm);
  gather_k<<<NPAD, 64, 0, stream>>>(hf, perm, hg);

  // ---- expert MLP (routed): l0 relu, l1 relu + skip(hg), l2 relu, l3 ----
  const int wstr = LAYERS * HH;   // f16 elements between experts per layer base
  const int bstr = LAYERS * HDIM;
  dim3 eg(NPAD / 64, HDIM / 64);
  gemm_wmma_f16<<<eg, 128, 0, stream>>>(hg, Wet + 0 * HH, be + 0 * HDIM, nullptr,
                                        t1, HDIM, HDIM, 1, padOff, wstr, bstr);
  gemm_wmma_f16<<<eg, 128, 0, stream>>>(t1, Wet + 1 * HH, be + 1 * HDIM, hg,
                                        t2, HDIM, HDIM, 1, padOff, wstr, bstr);
  gemm_wmma_f16<<<eg, 128, 0, stream>>>(t2, Wet + 2 * HH, be + 2 * HDIM, nullptr,
                                        t1, HDIM, HDIM, 1, padOff, wstr, bstr);
  gemm_wmma_f16<<<eg, 128, 0, stream>>>(t1, Wet + 3 * HH, be + 3 * HDIM, nullptr,
                                        t2, HDIM, HDIM, 0, padOff, wstr, bstr);
  scatter_k<<<NTOK, 64, 0, stream>>>(t2, posA, gatev, moe);

  // ---- sigma head ----
  gemm_wmma_f16<<<dim3(NTOK / 64, HEADD / 64), 128, 0, stream>>>(
      moe, Ws1t, bs1, nullptr, s1b, HDIM, HEADD, 1, nullptr, 0, 0);

  // ---- color head ----
  copy_moe_cin_k<<<NTOK, 64, 0, stream>>>(moe, cin);
  posenc_dir_k<<<NTOK / 256, 256, 0, stream>>>(dirs, cin);
  gemm_wmma_f16<<<dim3(NTOK / 64, HEADD / 64), 128, 0, stream>>>(
      cin, Wc1t, bc1, nullptr, c1b, CINP, HEADD, 1, nullptr, 0, 0);

  // ---- finals ----
  heads_k<<<NTOK / 256, 256, 0, stream>>>(s1b, c1b, Ws2, bs2, Wc2, bc2, out);
}
